// set_transofrm_77034533421807
// MI455X (gfx1250) — matrix operand order repaired and checked by reading, not compiled, not run on hardware
//
#include <hip/hip_runtime.h>
#include <hip/hip_bf16.h>

// ---- Problem constants -----------------------------------------------------
#define NSET    32
#define NITEM   128
#define NHEAD   8
#define HDIM    64
#define DMODEL  512
#define XROWS   (NSET * NITEM)          // 4096

typedef __bf16 bf16_t;
typedef __attribute__((ext_vector_type(16))) __bf16 v16bf;
typedef __attribute__((ext_vector_type(8)))  __bf16 v8bf;
typedef __attribute__((ext_vector_type(8)))  float   v8f;

#define WMMA_BF16(a, b, c) \
  __builtin_amdgcn_wmma_f32_16x16x32_bf16(false, (a), false, (b), (short)0, (c), false, false)

// ---- WMMA fragment loaders (layouts per CDNA5 ISA 7.12.2, wave32) ----------
// A-matrix 16x32 bf16: lane L holds row M = L%16.
//   lanes 0-15 : e0..7 = K[k..k+7],   e8..15 = K[k+16..k+23]
//   lanes 16-31: e0..7 = K[k+8..k+15],e8..15 = K[k+24..k+31]
__device__ __forceinline__ v16bf load_frag_a(const bf16_t* __restrict__ p, int ld,
                                             int m, int k, int lane) {
  int half = lane >> 4;
  const bf16_t* row = p + (size_t)(m + (lane & 15)) * ld + k;
  v8bf lo = *(const v8bf*)(row + half * 8);
  v8bf hi = *(const v8bf*)(row + 16 + half * 8);
  return __builtin_shufflevector(lo, hi, 0,1,2,3,4,5,6,7,8,9,10,11,12,13,14,15);
}

// B-matrix 32x16 bf16: lane L holds column N = L%16.
//   lanes 0-15 : e0..7 = K[k..k+7],   e8..15 = K[k+16..k+23]
//   lanes 16-31: e0..7 = K[k+8..k+15],e8..15 = K[k+24..k+31]
// Source is stored "N-major" (Bt[n][k]) so each lane reads two runs of 8 contiguous bf16.
__device__ __forceinline__ v16bf load_frag_b(const bf16_t* __restrict__ p, int ld,
                                             int n, int k, int lane) {
  int half = lane >> 4;
  const bf16_t* row = p + (size_t)(n + (lane & 15)) * ld + k;
  v8bf lo = *(const v8bf*)(row + half * 8);
  v8bf hi = *(const v8bf*)(row + 16 + half * 8);
  return __builtin_shufflevector(lo, hi, 0,1,2,3,4,5,6,7,8,9,10,11,12,13,14,15);
}

// ---- Kernel 1a: elementwise f32 -> bf16 ------------------------------------
__global__ void k_cvt_bf16(const float* __restrict__ in, bf16_t* __restrict__ out, int n) {
  int i = blockIdx.x * blockDim.x + threadIdx.x;
  if (i < n) out[i] = (bf16_t)in[i];
}

// ---- Kernel 1b: transpose + convert (+scale) f32 [rows,cols] -> bf16 [cols,rows]
__global__ void k_transpose_cvt(const float* __restrict__ in, bf16_t* __restrict__ out,
                                int rows, int cols, float scale) {
  int i = blockIdx.x * blockDim.x + threadIdx.x;
  if (i >= rows * cols) return;
  int r = i / cols, c = i - r * cols;
  out[(size_t)c * rows + r] = (bf16_t)(in[i] * scale);
}

// ---- Kernel 2: QKV projection  A[4096,512] @ W (Bt = Wt[512,512]) ----------
// Output layout [S, H, N, 64] bf16. grid = (16, 8), block = 256 (8 waves).
// Each wave computes a 32(M) x 64(N) tile: 2 A-frags x 4 B-frags per K-step.
__global__ void k_gemm_qkv(const bf16_t* __restrict__ A, const bf16_t* __restrict__ Bt,
                           bf16_t* __restrict__ out) {
  int lane = threadIdx.x & 31;
  int wave = threadIdx.x >> 5;
  int m_base = blockIdx.x * 256 + wave * 32;
  int n_base = blockIdx.y * 64;
  v8f acc[2][4] = {};
#pragma unroll 2
  for (int k = 0; k < DMODEL; k += 32) {
    v16bf a0 = load_frag_a(A, DMODEL, m_base,      k, lane);
    v16bf a1 = load_frag_a(A, DMODEL, m_base + 16, k, lane);
#pragma unroll
    for (int t = 0; t < 4; ++t) {
      v16bf b = load_frag_b(Bt, DMODEL, n_base + t * 16, k, lane);
      acc[0][t] = WMMA_BF16(a0, b, acc[0][t]);
      acc[1][t] = WMMA_BF16(a1, b, acc[1][t]);
    }
  }
  int half = lane >> 4, nl = lane & 15;
#pragma unroll
  for (int mt = 0; mt < 2; ++mt) {
#pragma unroll
    for (int t = 0; t < 4; ++t) {
      int c = n_base + t * 16 + nl;
      int h = c >> 6, d = c & 63;
#pragma unroll
      for (int r = 0; r < 8; ++r) {
        int m = m_base + mt * 16 + r + 8 * half;
        int s = m >> 7, nit = m & 127;
        out[((((size_t)s * NHEAD + h) * NITEM) + nit) * HDIM + d] = (bf16_t)acc[mt][t][r];
      }
    }
  }
}

// ---- Kernel 3: vw[s,h] = v[s,h](128x64) @ W4_h(64x64), stored transposed ---
// V: [S*H][128][64], W4t: [64][512], out VWt: [S*H][64][128].
// 256 blocks, 128 threads (4 waves), wave tile = 32(M) x 64(N).
__global__ void k_gemm_vw(const bf16_t* __restrict__ V, const bf16_t* __restrict__ W4t,
                          bf16_t* __restrict__ VWt) {
  int sh = blockIdx.x;
  int h  = sh & 7;
  int lane = threadIdx.x & 31, wave = threadIdx.x >> 5;
  const bf16_t* Av = V + (size_t)sh * NITEM * HDIM;
  bf16_t* out = VWt + (size_t)sh * HDIM * NITEM;
  int m_base = wave * 32;
  v8f acc[2][4] = {};
#pragma unroll
  for (int k = 0; k < HDIM; k += 32) {
    v16bf a0 = load_frag_a(Av, HDIM, m_base,      k, lane);
    v16bf a1 = load_frag_a(Av, HDIM, m_base + 16, k, lane);
#pragma unroll
    for (int t = 0; t < 4; ++t) {
      v16bf b = load_frag_b(W4t, DMODEL, t * 16, h * HDIM + k, lane);
      acc[0][t] = WMMA_BF16(a0, b, acc[0][t]);
      acc[1][t] = WMMA_BF16(a1, b, acc[1][t]);
    }
  }
  int half = lane >> 4, nl = lane & 15;
#pragma unroll
  for (int mt = 0; mt < 2; ++mt) {
#pragma unroll
    for (int t = 0; t < 4; ++t) {
      int n = t * 16 + nl;
      v8bf pack;
#pragma unroll
      for (int r = 0; r < 8; ++r) pack[r] = (bf16_t)acc[mt][t][r];
      // transposed store: 8 consecutive m at row n (16B aligned)
      *(v8bf*)(out + (size_t)n * NITEM + m_base + mt * 16 + 8 * half) = pack;
    }
  }
}

// ---- Kernel 4: attn[s,h] = relu(q @ k^T) / N  (1/sqrt(D) folded into q) ----
// Q,K: [S*H][128][64], out Attn: [S*H][128][128]. 256 blocks, 8 waves.
// Wave tile = 32(M) x 64(N): waves arranged 4(M) x 2(N).
__global__ void k_gemm_attn(const bf16_t* __restrict__ Q, const bf16_t* __restrict__ Kt,
                            bf16_t* __restrict__ Attn) {
  int sh = blockIdx.x;
  int lane = threadIdx.x & 31, wave = threadIdx.x >> 5;
  const bf16_t* q  = Q  + (size_t)sh * NITEM * HDIM;
  const bf16_t* kk = Kt + (size_t)sh * NITEM * HDIM;   // B[k][n] = k[n][k]: row-contig
  bf16_t* out = Attn + (size_t)sh * NITEM * NITEM;
  int m_base = (wave >> 1) * 32;
  int n_base = (wave & 1) * 64;
  v8f acc[2][4] = {};
#pragma unroll
  for (int k = 0; k < HDIM; k += 32) {
    v16bf a0 = load_frag_a(q, HDIM, m_base,      k, lane);
    v16bf a1 = load_frag_a(q, HDIM, m_base + 16, k, lane);
#pragma unroll
    for (int t = 0; t < 4; ++t) {
      v16bf b = load_frag_b(kk, HDIM, n_base + t * 16, k, lane);
      acc[0][t] = WMMA_BF16(a0, b, acc[0][t]);
      acc[1][t] = WMMA_BF16(a1, b, acc[1][t]);
    }
  }
  int half = lane >> 4, nl = lane & 15;
  const float inv_n = 1.0f / (float)NITEM;
#pragma unroll
  for (int mt = 0; mt < 2; ++mt) {
#pragma unroll
    for (int t = 0; t < 4; ++t) {
      int n = n_base + t * 16 + nl;
#pragma unroll
      for (int r = 0; r < 8; ++r) {
        int m = m_base + mt * 16 + r + 8 * half;
        float v = acc[mt][t][r];
        v = v > 0.0f ? v * inv_n : 0.0f;
        out[(size_t)m * NITEM + n] = (bf16_t)v;
      }
    }
  }
}

// ---- Kernel 5: out[i,j] = sum_h attn[j,h](128x128) @ vw[i,h](128x64) -------
// Attn: [S][H][128][128], VWt: [S][H][64][128], Out: [S][S][128][64] f32.
// 1024 blocks (pair = i*32 + j), 128 threads (4 waves), wave tile 32x64.
// K = 8 heads * 128 = 1024.
__global__ void k_gemm_out(const bf16_t* __restrict__ Attn, const bf16_t* __restrict__ VWt,
                           float* __restrict__ Out) {
  int pair = blockIdx.x;
  int i = pair >> 5, j = pair & 31;
  int lane = threadIdx.x & 31, wave = threadIdx.x >> 5;
  const bf16_t* Aj = Attn + (size_t)j * NHEAD * NITEM * NITEM;
  const bf16_t* Bi = VWt  + (size_t)i * NHEAD * HDIM * NITEM;
  int m_base = wave * 32;
  v8f acc[2][4] = {};
  for (int h = 0; h < NHEAD; ++h) {
    const bf16_t* Ah = Aj + (size_t)h * NITEM * NITEM;
    const bf16_t* Bh = Bi + (size_t)h * HDIM * NITEM;
#pragma unroll
    for (int k = 0; k < NITEM; k += 32) {
      v16bf a0 = load_frag_a(Ah, NITEM, m_base,      k, lane);
      v16bf a1 = load_frag_a(Ah, NITEM, m_base + 16, k, lane);
#pragma unroll
      for (int t = 0; t < 4; ++t) {
        v16bf b = load_frag_b(Bh, NITEM, t * 16, k, lane);
        acc[0][t] = WMMA_BF16(a0, b, acc[0][t]);
        acc[1][t] = WMMA_BF16(a1, b, acc[1][t]);
      }
    }
  }
  int half = lane >> 4, nl = lane & 15;
  float* o = Out + (size_t)pair * NITEM * HDIM;
#pragma unroll
  for (int mt = 0; mt < 2; ++mt) {
#pragma unroll
    for (int t = 0; t < 4; ++t) {
#pragma unroll
      for (int r = 0; r < 8; ++r) {
        o[(size_t)(m_base + mt * 16 + r + 8 * half) * HDIM + t * 16 + nl] = acc[mt][t][r];
      }
    }
  }
}

// ---- Host launch -----------------------------------------------------------
extern "C" void kernel_launch(void* const* d_in, const int* in_sizes, int n_in,
                              void* d_out, int out_size, void* d_ws, size_t ws_size,
                              hipStream_t stream) {
  const float* x  = (const float*)d_in[0];   // [32,128,512]
  const float* y  = (const float*)d_in[1];   // [32,128,512]
  const float* W1 = (const float*)d_in[2];   // [512,512]
  const float* W2 = (const float*)d_in[3];   // [512,512]
  const float* W3 = (const float*)d_in[4];   // [512,512]
  const float* W4 = (const float*)d_in[5];   // [512,64]
  float* out = (float*)d_out;                // [32,32,128,64]

  char* ws = (char*)d_ws;
  const size_t MB = 1024 * 1024;
  bf16_t* xb   = (bf16_t*)(ws + 0 * MB);     // [4096,512]   4 MB
  bf16_t* yb   = (bf16_t*)(ws + 4 * MB);     // [4096,512]   4 MB
  bf16_t* w1t  = (bf16_t*)(ws + 8 * MB);     // [512,512]    0.5 MB (scaled 1/8)
  bf16_t* w2t  = (bf16_t*)(ws + 8 * MB + 512 * 1024);
  bf16_t* w3t  = (bf16_t*)(ws + 9 * MB);
  bf16_t* w4t  = (bf16_t*)(ws + 9 * MB + 512 * 1024);   // [64,512] 64 KB
  bf16_t* qb   = (bf16_t*)(ws + 10 * MB);    // [S,H,128,64] 4 MB
  bf16_t* kb   = (bf16_t*)(ws + 14 * MB);    // [S,H,128,64] 4 MB
  bf16_t* vb   = (bf16_t*)(ws + 18 * MB);    // [S,H,128,64] 4 MB
  bf16_t* vwt  = (bf16_t*)(ws + 22 * MB);    // [S,H,64,128] 4 MB
  bf16_t* attn = (bf16_t*)(ws + 26 * MB);    // [S,H,128,128] 8 MB  (total 34 MB)

  const int nXY = XROWS * DMODEL;            // 2,097,152
  const float inv_sqrt_d = 0.125f;           // 1/sqrt(64), folded into W1^T

  // Stage 1: convert / transpose weights and activations to bf16
  k_cvt_bf16<<<(nXY + 255) / 256, 256, 0, stream>>>(x, xb, nXY);
  k_cvt_bf16<<<(nXY + 255) / 256, 256, 0, stream>>>(y, yb, nXY);
  k_transpose_cvt<<<(DMODEL * DMODEL + 255) / 256, 256, 0, stream>>>(W1, w1t, DMODEL, DMODEL, inv_sqrt_d);
  k_transpose_cvt<<<(DMODEL * DMODEL + 255) / 256, 256, 0, stream>>>(W2, w2t, DMODEL, DMODEL, 1.0f);
  k_transpose_cvt<<<(DMODEL * DMODEL + 255) / 256, 256, 0, stream>>>(W3, w3t, DMODEL, DMODEL, 1.0f);
  k_transpose_cvt<<<(DMODEL * HDIM + 255) / 256, 256, 0, stream>>>(W4, w4t, DMODEL, HDIM, 1.0f);

  // Stage 2: QKV projections (q pre-scaled by 1/sqrt(D))
  dim3 qkv_grid(XROWS / 256, DMODEL / 64);
  k_gemm_qkv<<<qkv_grid, 256, 0, stream>>>(xb, w1t, qb);
  k_gemm_qkv<<<qkv_grid, 256, 0, stream>>>(yb, w2t, kb);
  k_gemm_qkv<<<qkv_grid, 256, 0, stream>>>(yb, w3t, vb);

  // Stage 3: vw[s,h] = v[s,h] @ W4_h  (stored transposed for B-fragment reads)
  k_gemm_vw<<<NSET * NHEAD, 128, 0, stream>>>(vb, w4t, vwt);

  // Stage 4: attn = relu(q @ k^T) / N
  k_gemm_attn<<<NSET * NHEAD, 256, 0, stream>>>(qb, kb, attn);

  // Stage 5: out[i,j] = sum_h attn[j,h] @ vw[i,h]
  k_gemm_out<<<NSET * NSET, 128, 0, stream>>>(attn, vwt, out);
}
